// NeuralODE_18700287607296
// MI455X (gfx1250) — matrix operand order repaired and checked by reading, not compiled, not run on hardware
//
#include <hip/hip_runtime.h>
#include <hip/hip_bf16.h>

typedef __attribute__((ext_vector_type(16))) __bf16 v16bf;
typedef __attribute__((ext_vector_type(8)))  __bf16 v8bf;
typedef __attribute__((ext_vector_type(8)))  float  v8f;

#define TT   256      // time steps
#define NW   128      // hidden width
#define NS   2        // state dim
#define AST  136      // padded activation row stride (bf16 elems) -> 272B, bank-friendly

#define LOG2E 1.4426950408889634f
#define LN2   0.6931471805599453f

// ---------------- helpers ----------------

__device__ __forceinline__ float softplus_f(float x) {
    // softplus(x) = max(x,0) + log1p(exp(-|x|)) with bare hardware trans ops
    // (v_exp_f32 = 2^x, v_log_f32 = log2 x).  Log argument is in (1,2], so no
    // denormal guards are needed; result is truncated to bf16 afterwards.
    const float z = __builtin_amdgcn_exp2f(-LOG2E * fabsf(x));
    return fmaxf(x, 0.f) + LN2 * __builtin_amdgcn_logf(1.f + z);
}

// Load a 16x32 bf16 A-fragment (M x K) from an LDS activation buffer laid out
// row-major [16][stride].  Contiguous k order, same as the B packing: lane holds row m=lane&15,
// K = 16*hb + [0..15]  (hb = lane>>4), 2 elems/VGPR.
__device__ __forceinline__ v16bf load_afrag(const __bf16* buf, int stride, int kc) {
    const int lane = threadIdx.x & 31;
    const int r  = lane & 15;
    const int hb = lane >> 4;
    const __bf16* p = buf + r * stride + kc * 32 + hb * 16;
    v16bf a;
    *(v8bf*)&a        = *(const v8bf*)(p);
    *((v8bf*)&a + 1)  = *(const v8bf*)(p + 8);
    return a;
}

// D (16x16 f32) layout: lane covers column n=lane&15, VGPR v holds row m=v+8*(lane>>4).
// Apply bias + softplus, convert to bf16, store into activation buffer.
__device__ __forceinline__ void store_act(__bf16* buf, v8f acc, int nt, float bs) {
    const int lane  = threadIdx.x & 31;
    const int col   = lane & 15;
    const int hb    = lane >> 4;
    const int nglob = nt * 16 + col;
#pragma unroll
    for (int v = 0; v < 8; ++v) {
        const int m = v + 8 * hb;
        buf[m * AST + nglob] = (__bf16)softplus_f(acc[v] + bs);
    }
}

// ---------------- kernel ----------------

__global__ __launch_bounds__(128, 1)
void node_rk6_wmma(const float* __restrict__ ts, const float* __restrict__ y0,
                   const float* __restrict__ us, const float* __restrict__ w1,
                   const float* __restrict__ b1, const float* __restrict__ w2,
                   const float* __restrict__ b2, const float* __restrict__ w3,
                   const float* __restrict__ b3, float* __restrict__ out)
{
    // Weight B-fragments, packed in exact WMMA lane layout (lane -> 16 contiguous bf16).
    __shared__ v16bf w1f[8][32];        //  8 KB  (K padded 3->32)
    __shared__ v16bf w2f[8][4][32];     // 32 KB
    __shared__ v16bf w3f[4][32];        //  4 KB  (N padded 2->16)
    __shared__ __align__(16) __bf16 xbuf [16 * 32];   // stage input  [16 rows][K pad 32]
    __shared__ __align__(16) __bf16 h1buf[16 * AST];  // hidden 1
    __shared__ __align__(16) __bf16 h2buf[16 * AST];  // hidden 2
    __shared__ float ybuf[32];                        // y  [16 rows][2]
    __shared__ float kbuf[6][32];                     // k1..k6

    const int tid  = threadIdx.x;
    const int lane = tid & 31;
    const int wv   = tid >> 5;          // wave 0..3
    const int b0   = blockIdx.x * 16;   // batch tile start

    // ---- pack w2 fragments: B[k][n] = w2[n][k] ----
    for (int idx = tid; idx < 8 * 4 * 32 * 16; idx += blockDim.x) {
        const int e  = idx & 15;
        const int ln = (idx >> 4) & 31;
        const int kc = (idx >> 9) & 3;
        const int nt = idx >> 11;
        const int j  = ln & 15, kh = ln >> 4;
        const int k  = kc * 32 + kh * 16 + e;
        const int n  = nt * 16 + j;
        ((__bf16*)w2f)[idx] = (__bf16)w2[n * NW + k];
    }
    // ---- pack w1 fragments (K padded: k>=3 -> 0) ----
    for (int idx = tid; idx < 8 * 32 * 16; idx += blockDim.x) {
        const int e  = idx & 15;
        const int ln = (idx >> 4) & 31;
        const int nt = idx >> 9;
        const int j  = ln & 15, kh = ln >> 4;
        const int k  = kh * 16 + e;
        const int n  = nt * 16 + j;
        ((__bf16*)w1f)[idx] = (k < 3) ? (__bf16)w1[n * 3 + k] : (__bf16)0.f;
    }
    // ---- pack w3 fragments (N padded: n>=2 -> 0) ----
    for (int idx = tid; idx < 4 * 32 * 16; idx += blockDim.x) {
        const int e  = idx & 15;
        const int ln = (idx >> 4) & 31;
        const int kc = idx >> 9;
        const int j  = ln & 15, kh = ln >> 4;
        const int k  = kc * 32 + kh * 16 + e;
        ((__bf16*)w3f)[idx] = (j < NS) ? (__bf16)w3[j * NW + k] : (__bf16)0.f;
    }
    for (int idx = tid; idx < 16 * 32; idx += blockDim.x) xbuf[idx] = (__bf16)0.f;

    // ---- loop-invariant per-lane bias registers ----
    const int colA = (wv * 2 + 0) * 16 + (lane & 15);
    const int colB = (wv * 2 + 1) * 16 + (lane & 15);
    const float b1A = b1[colA], b1B = b1[colB];
    const float b2A = b2[colA], b2B = b2[colB];
    float b3r = 0.f;
    if (wv == 0 && (lane & 15) < NS) b3r = b3[lane & 15];

    float tcur = ts[0];
    float u0reg = 0.f;
    if (tid < 16) u0reg = us[(size_t)(b0 + tid) * TT + 0];   // us[:,0] (mid stages)
    if (tid < 32) {
        const int r = tid >> 1, st = tid & 1;
        const float v = y0[(size_t)(b0 + r) * NS + st];
        ybuf[tid] = v;
        out[(size_t)(b0 + r) * TT * NS + st] = v;            // ys[:,0,:] = y0
        xbuf[r * 32 + st] = (__bf16)v;                       // stage-0 input, step 0
    }
    if (tid < 16) xbuf[tid * 32 + 2] = (__bf16)u0reg;        // u for stage 0, step 0
    __syncthreads();

    // RK stage matrix (lower triangle); row s used to build the input of stage s
    const float SA[6][5] = {
        { 0.f, 0.f, 0.f, 0.f, 0.f },
        { 0.161f, 0.f, 0.f, 0.f, 0.f },
        { -0.008480655492356989f, 0.335480655492357f, 0.f, 0.f, 0.f },
        { 2.8971530571054935f, -6.359448489975075f, 4.3622954328695815f, 0.f, 0.f },
        { 5.325864828439257f, -11.748883564062828f, 7.4955393428898365f, -0.09249506636175525f, 0.f },
        { 5.86145544294642f, -12.92096931784711f, 8.159367898576159f, -0.071584973281401f, -0.028269050394068383f }
    };

    for (int i = 0; i < TT - 1; ++i) {
        // Prefetch: used at s=4/5 prep (us) and in all stage combinations (ts),
        // several barriers after issue -> latency fully hidden.
        const float tnext = ts[i + 1];
        float unext = 0.f;
        if (tid < 16) unext = us[(size_t)(b0 + tid) * TT + i + 1];
        const float hh = tnext - tcur;
        tcur = tnext;

#pragma unroll
        for (int s = 0; s < 6; ++s) {
            // On entry: xbuf holds the stage-s MLP input (prepared last region).

            // ---- layer 1: softplus(x @ w1^T + b1) ----
            {
                const v16bf a = load_afrag(xbuf, 32, 0);
#pragma unroll
                for (int q = 0; q < 2; ++q) {
                    const int nt = wv * 2 + q;
                    v8f acc = {};
                    acc = __builtin_amdgcn_wmma_f32_16x16x32_bf16(
                        false, a, false, w1f[nt][lane], (short)0, acc, false, false);
                    store_act(h1buf, acc, nt, q == 0 ? b1A : b1B);
                }
            }
            __syncthreads();

            // ---- layer 2: softplus(h1 @ w2^T + b2) ----
#pragma unroll
            for (int q = 0; q < 2; ++q) {
                const int nt = wv * 2 + q;
                v8f acc = {};
#pragma unroll
                for (int kc = 0; kc < 4; ++kc) {
                    const v16bf a = load_afrag(h1buf, AST, kc);
                    acc = __builtin_amdgcn_wmma_f32_16x16x32_bf16(
                        false, a, false, w2f[nt][kc][lane], (short)0, acc, false, false);
                }
                store_act(h2buf, acc, nt, q == 0 ? b2A : b2B);
            }
            __syncthreads();

            // ---- layer 3 + prep of the next stage input (wave 0 only) ----
            if (wv == 0) {
                // Two accumulators halve the WMMA->WMMA RAW chain.
                v8f acc0 = {}, acc1 = {};
                {
                    const v16bf a0 = load_afrag(h2buf, AST, 0);
                    acc0 = __builtin_amdgcn_wmma_f32_16x16x32_bf16(
                        false, a0, false, w3f[0][lane], (short)0, acc0, false, false);
                    const v16bf a1 = load_afrag(h2buf, AST, 1);
                    acc1 = __builtin_amdgcn_wmma_f32_16x16x32_bf16(
                        false, a1, false, w3f[1][lane], (short)0, acc1, false, false);
                    const v16bf a2 = load_afrag(h2buf, AST, 2);
                    acc0 = __builtin_amdgcn_wmma_f32_16x16x32_bf16(
                        false, a2, false, w3f[2][lane], (short)0, acc0, false, false);
                    const v16bf a3 = load_afrag(h2buf, AST, 3);
                    acc1 = __builtin_amdgcn_wmma_f32_16x16x32_bf16(
                        false, a3, false, w3f[3][lane], (short)0, acc1, false, false);
                }
                const int col = lane & 15, hb = lane >> 4;
                if (col < NS) {
#pragma unroll
                    for (int v = 0; v < 8; ++v) {
                        const int m = v + 8 * hb;
                        kbuf[s][m * 2 + col] = acc0[v] + acc1[v] + b3r;
                    }
                }
            }
            // Folded prep: build xbuf for stage s+1 (or stage 0 of step i+1).
            if (tid < 32) {
                float v;
                if (s < 5) {
                    v = ybuf[tid];
#pragma unroll
                    for (int j = 0; j < 5; ++j)
                        if (j <= s) v = fmaf(hh * SA[s + 1][j], kbuf[j][tid], v);
                } else {
                    const float a = 0.09646076681806523f  * kbuf[0][tid]
                                  + 0.01f                 * kbuf[1][tid]
                                  + 0.4798896504144996f   * kbuf[2][tid]
                                  + 1.379008574103742f    * kbuf[3][tid]
                                  + (-3.290069515436081f) * kbuf[4][tid]
                                  + 2.324710524099774f    * kbuf[5][tid];
                    v = fmaf(hh, a, ybuf[tid]);
                    ybuf[tid] = v;
                    const int r = tid >> 1, st = tid & 1;
                    out[(size_t)(b0 + r) * TT * NS + (size_t)(i + 1) * NS + st] = v;
                }
                const int r = tid >> 1, st = tid & 1;
                xbuf[r * 32 + st] = (__bf16)v;
            }
            if (tid < 16) {
                // u of stage s+1: stages 1..4 -> us[:,0]; stage 5 -> us[:,i+1];
                // stage 0 of step i+1 -> us[:,i+1] as well (k1 input).
                const float u = (s >= 4) ? unext : u0reg;
                xbuf[tid * 32 + 2] = (__bf16)u;
            }
            __syncthreads();
        }
    }
}

extern "C" void kernel_launch(void* const* d_in, const int* in_sizes, int n_in,
                              void* d_out, int out_size, void* d_ws, size_t ws_size,
                              hipStream_t stream) {
    const float* ts = (const float*)d_in[0];
    const float* y0 = (const float*)d_in[1];
    const float* us = (const float*)d_in[2];
    const float* w1 = (const float*)d_in[3];
    const float* b1 = (const float*)d_in[4];
    const float* w2 = (const float*)d_in[5];
    const float* b2 = (const float*)d_in[6];
    const float* w3 = (const float*)d_in[7];
    const float* b3 = (const float*)d_in[8];
    float* out = (float*)d_out;

    const int B = in_sizes[1] / NS;      // y0 is (B, STATE)
    dim3 grid(B / 16), block(128);
    node_rk6_wmma<<<grid, block, 0, stream>>>(ts, y0, us, w1, b1, w2, b2, w3, b3, out);
}
